// ProteinGraphModel_56453050138691
// MI455X (gfx1250) — matrix operand order repaired and checked by reading, not compiled, not run on hardware
//
#include <hip/hip_runtime.h>
#include <hip/hip_bf16.h>

typedef _Float16 half_t;
typedef __attribute__((ext_vector_type(16))) _Float16 v16h;
typedef __attribute__((ext_vector_type(8)))  _Float16 v8h;
typedef __attribute__((ext_vector_type(8)))  float    v8f;

// ---------------------------------------------------------------------------
// WMMA GEMM with register blocking along N (NT n-tiles per wave; A fragment
// loaded once per k-block and reused for NT WMMAs).
//   A: f16 [M x lda], lda multiple of 8, K-extent = kblocks*32 (zero padded)
//   Wp: pre-packed B fragments, layout [ntile][kblock][lane][16 halfs]
//   M multiple of 128 (grid covers exactly; EXEC all-1 for WMMA)
// OUT_MODE: 0 = f16 store (full tile, zeros past Nout)
//           1 = f32 store (col < Nout)
//           2 = f32 atomic scatter-add at row scat_idx[row] (col < Nout)
// ---------------------------------------------------------------------------
template<bool RELU, int OUT_MODE, int NT>
__global__ void wmma_gemm(const half_t* __restrict__ A, int lda,
                          const half_t* __restrict__ Wp, const float* __restrict__ bias,
                          void* __restrict__ Cv, int ldc,
                          const int* __restrict__ scat_idx,
                          int kblocks, int Nout)
{
    const int lane = threadIdx.x & 31;
    const int wave = threadIdx.x >> 5;
    const int m0   = (blockIdx.x * 8 + wave) * 16;
    const int nt0  = blockIdx.y * NT;

    const int rowA   = m0 + (lane & 15);
    const int kbaseA = (lane < 16) ? 0 : 8;        // A 16x32 f16 lane layout

    const half_t* ap = A + (size_t)rowA * lda + kbaseA;
    const half_t* bp = Wp + ((size_t)nt0 * kblocks * 32 + lane) * 16;
    const size_t  bstride = (size_t)kblocks * 512; // halfs between n-tiles

    v8f acc[NT];
#pragma unroll
    for (int i = 0; i < NT; ++i) acc[i] = (v8f){};

    for (int kb = 0; kb < kblocks; ++kb) {
        v8h alo = *(const v8h*)(ap);               // K 0..7   (or 8..15)
        v8h ahi = *(const v8h*)(ap + 16);          // K 16..23 (or 24..31)
        v16h a = __builtin_shufflevector(alo, ahi, 0,1,2,3,4,5,6,7,8,9,10,11,12,13,14,15);
#pragma unroll
        for (int nt = 0; nt < NT; ++nt) {
            const half_t* bq = bp + (size_t)nt * bstride;
            v8h b0 = *(const v8h*)(bq);
            v8h b1 = *(const v8h*)(bq + 8);
            v16h b = __builtin_shufflevector(b0, b1, 0,1,2,3,4,5,6,7,8,9,10,11,12,13,14,15);
            acc[nt] = __builtin_amdgcn_wmma_f32_16x16x32_f16(false, a, false, b,
                                                             (short)0, acc[nt], false, false);
        }
        ap += 32;
        bp += 512;                                 // next k-block
    }

#pragma unroll
    for (int nt = 0; nt < NT; ++nt) {
        const int col = (nt0 + nt) * 16 + (lane & 15);
        const float bv = (bias && col < Nout) ? bias[col] : 0.f;
#pragma unroll
        for (int r = 0; r < 8; ++r) {
            int row = m0 + r + ((lane < 16) ? 0 : 8);   // C/D VGPR layout
            float v = acc[nt][r] + bv;                  // pad cols: acc==0, bv==0
            if (RELU) v = fmaxf(v, 0.f);
            if (OUT_MODE == 0) {
                ((half_t*)Cv)[(size_t)row * ldc + col] = (half_t)v;
            } else if (OUT_MODE == 1) {
                if (col < Nout) ((float*)Cv)[(size_t)row * ldc + col] = v;
            } else {
                if (col < Nout) atomicAdd(&((float*)Cv)[(size_t)scat_idx[row] * ldc + col], v);
            }
        }
    }
}

// Pack W[Kd x Nout] (f32) into B-fragment layout: ((nt*kblocks+kb)*32+lane)*16+j
__global__ void k_pack_w(const float* __restrict__ W, half_t* __restrict__ Wp,
                         int Kd, int Nout, int kblocks, long total)
{
    long t = (long)blockIdx.x * blockDim.x + threadIdx.x;
    if (t >= total) return;
    int  j    = (int)(t & 15);
    int  lane = (int)((t >> 4) & 31);
    long blk  = t >> 9;
    int  kb   = (int)(blk % kblocks);
    int  nt   = (int)(blk / kblocks);
    int  col  = nt * 16 + (lane & 15);
    int  k    = kb * 32 + ((lane < 16) ? 0 : 8) + (j & 7) + ((j >> 3) << 4);
    Wp[t] = (col < Nout && k < Kd) ? (half_t)W[(size_t)k * Nout + col] : (half_t)0.f;
}

// ---------------------------------------------------------------------------
// Elementwise / gather / scatter / norm kernels (all producers zero-fill pads)
// ---------------------------------------------------------------------------
__global__ void k_build_fa(const float* __restrict__ emb, const int* __restrict__ idx,
                           const float* __restrict__ dist, half_t* __restrict__ fa, long total)
{
    long t = (long)blockIdx.x * blockDim.x + threadIdx.x;
    if (t >= total) return;
    long i = t >> 6; int c = (int)(t & 63);        // width 64 (33 real)
    float v = (c < 32) ? emb[(size_t)idx[i] * 32 + c]
            : (c == 32) ? (1.f / dist[i]) : 0.f;
    fa[t] = (half_t)v;
}

__global__ void k_ksum(const half_t* __restrict__ aA, half_t* __restrict__ out, long total)
{
    long t = (long)blockIdx.x * blockDim.x + threadIdx.x;
    if (t >= total) return;
    long n = t >> 5; int c = (int)(t & 31);        // out width 32 (15 real)
    float s = 0.f;
    if (c < 15) {
#pragma unroll
        for (int k = 0; k < 8; ++k) s += (float)aA[((size_t)n * 8 + k) * 16 + c]; // aA lda 16
    }
    out[t] = (half_t)s;
}

__global__ void k_concat16(const half_t* __restrict__ aB, const float* __restrict__ nx,
                           half_t* __restrict__ out, long total)
{
    long t = (long)blockIdx.x * blockDim.x + threadIdx.x;
    if (t >= total) return;
    long n = t >> 5; int c = (int)(t & 31);        // out width 32 (16 real)
    float v = (c < 15) ? (float)aB[(size_t)n * 16 + c]   // aB lda 16
            : (c == 15) ? nx[n] : 0.f;
    out[t] = (half_t)v;
}

__global__ void k_edge_dist(const float* __restrict__ pos, const int* __restrict__ src,
                            const int* __restrict__ dst, float* __restrict__ d, int ne)
{
    int e = blockIdx.x * blockDim.x + threadIdx.x;
    if (e >= ne) return;
    size_t s = (size_t)src[e] * 3, t = (size_t)dst[e] * 3;
    float dx = pos[s] - pos[t], dy = pos[s+1] - pos[t+1], dz = pos[s+2] - pos[t+2];
    d[e] = sqrtf(dx * dx + dy * dy + dz * dz);
}

__global__ void k_build_ein(const float* __restrict__ d, const float* __restrict__ ea,
                            const float* __restrict__ feat, const int* __restrict__ dst,
                            half_t* __restrict__ ein, long total)
{
    long t = (long)blockIdx.x * blockDim.x + threadIdx.x;
    if (t >= total) return;
    long e = t / 96; int c = (int)(t % 96);        // width 96 (82 real)
    float v = (c == 0) ? d[e] : (c == 1) ? ea[e]
            : (c < 82) ? feat[(size_t)dst[e] * 80 + (c - 2)] : 0.f;
    ein[t] = (half_t)v;
}

__global__ void k_zero(float* __restrict__ p, long n)
{
    long t = (long)blockIdx.x * blockDim.x + threadIdx.x;
    if (t < n) p[t] = 0.f;
}

__global__ void k_add(const float* __restrict__ a, const float* __restrict__ b,
                      float* __restrict__ o, long n)
{
    long t = (long)blockIdx.x * blockDim.x + threadIdx.x;
    if (t < n) o[t] = a[t] + b[t];
}

// o (f16, width Ds) = a + b (f32, width Din), zero pad columns
__global__ void k_add_cvt(const float* __restrict__ a, const float* __restrict__ b,
                          half_t* __restrict__ o, int Din, int Ds, long total)
{
    long t = (long)blockIdx.x * blockDim.x + threadIdx.x;
    if (t >= total) return;
    long n = t / Ds; int c = (int)(t % Ds);
    o[t] = (c < Din) ? (half_t)(a[(size_t)n * Din + c] + b[(size_t)n * Din + c]) : (half_t)0.f;
}

__global__ void k_cvt_pad(const float* __restrict__ a, half_t* __restrict__ o,
                          int Din, int Ds, long total)
{
    long t = (long)blockIdx.x * blockDim.x + threadIdx.x;
    if (t >= total) return;
    long n = t / Ds; int c = (int)(t % Ds);
    o[t] = (c < Din) ? (half_t)a[(size_t)n * Din + c] : (half_t)0.f;
}

__global__ void k_copy(const float* __restrict__ a, float* __restrict__ o, long n)
{
    long t = (long)blockIdx.x * blockDim.x + threadIdx.x;
    if (t < n) o[t] = a[t];
}

// out[dst[e]*D + c] += in[src[e]*D + c]
__global__ void k_scatter_add(const float* __restrict__ in, const int* __restrict__ src,
                              const int* __restrict__ dst, float* __restrict__ out,
                              int D, long total)
{
    long t = (long)blockIdx.x * blockDim.x + threadIdx.x;
    if (t >= total) return;
    long e = t / D; int c = (int)(t % D);
    atomicAdd(&out[(size_t)dst[e] * D + c], in[(size_t)src[e] * D + c]);
}

// row-wise layernorm + relu; blockDim = 256, D <= 256
// outHalf: store f16 with stride Ds (zero pad cols [D,Ds)); else f32 stride D
__global__ void k_ln_relu(const float* __restrict__ X, const float* __restrict__ gam,
                          const float* __restrict__ bet, void* __restrict__ out,
                          int D, int Ds, int outHalf)
{
    __shared__ float sm[256];
    int row = blockIdx.x, tid = threadIdx.x;
    float v = (tid < D) ? X[(size_t)row * D + tid] : 0.f;
    sm[tid] = v; __syncthreads();
    for (int s = 128; s > 0; s >>= 1) { if (tid < s) sm[tid] += sm[tid + s]; __syncthreads(); }
    float mean = sm[0] / D;
    __syncthreads();
    float dv = (tid < D) ? (v - mean) : 0.f;
    sm[tid] = dv * dv; __syncthreads();
    for (int s = 128; s > 0; s >>= 1) { if (tid < s) sm[tid] += sm[tid + s]; __syncthreads(); }
    float inv = rsqrtf(sm[0] / D + 1e-5f);
    float y = fmaxf((v - mean) * inv * ((tid < D) ? gam[tid] : 0.f)
                    + ((tid < D) ? bet[tid] : 0.f), 0.f);
    if (outHalf) {
        if (tid < Ds) ((half_t*)out)[(size_t)row * Ds + tid] = (tid < D) ? (half_t)y : (half_t)0.f;
    } else {
        if (tid < D) ((float*)out)[(size_t)row * D + tid] = y;
    }
}

// ---------------------------------------------------------------------------
// Host side
// ---------------------------------------------------------------------------
static inline dim3 g1d(long n) { return dim3((unsigned)((n + 255) / 256)); }

struct PW { const half_t* p; int kblocks; };

static void gemm(hipStream_t s, const half_t* A, int lda, PW w, const float* bias,
                 void* C, int ldc, const int* idx, int M, int Nout, int Nstore,
                 bool relu, int mode)
{
    const int NT = (Nstore == 16) ? 1 : (Nstore == 32) ? 2 : (Nstore == 80) ? 5
                 : (Nstore == 96) ? 6 : 8;
    dim3 g((unsigned)(M / 128), (unsigned)(Nstore / (16 * NT)));
    dim3 b(256);
#define GO(R, Md, T) wmma_gemm<R, Md, T><<<g, b, 0, s>>>(A, lda, w.p, bias, C, ldc, idx, w.kblocks, Nout)
#define SEL(R, Md) do { \
        if (NT == 1) GO(R, Md, 1); else if (NT == 2) GO(R, Md, 2); \
        else if (NT == 5) GO(R, Md, 5); else if (NT == 6) GO(R, Md, 6); \
        else GO(R, Md, 8); } while (0)
    if (mode == 0)      { if (relu) SEL(true, 0); else SEL(false, 0); }
    else if (mode == 1) { if (relu) SEL(true, 1); else SEL(false, 1); }
    else                { SEL(false, 2); }
#undef SEL
#undef GO
}

extern "C" void kernel_launch(void* const* d_in, const int* in_sizes, int n_in,
                              void* d_out, int out_size, void* d_ws, size_t ws_size,
                              hipStream_t stream)
{
    (void)in_sizes; (void)n_in; (void)out_size; (void)ws_size;
    const int N = 65536, NE = 524288;
    const long NK = 524288;           // N*K

    const float* pos       = (const float*)d_in[0];
    const float* node_x    = (const float*)d_in[1];
    const float* atom_dist = (const float*)d_in[2];
    const int*   atom_idx  = (const int*)d_in[3];
    const int*   e_src     = (const int*)d_in[4];
    const int*   e_dst     = e_src + NE;
    const float* edge_attr = (const float*)d_in[5];
    const float* emb       = (const float*)d_in[6];

    // ---- workspace layout ----
    char* ws = (char*)d_ws;
    size_t o = 0;
    auto take = [&](size_t bytes) { size_t r = o; o += (bytes + 255) & ~(size_t)255; return r; };
    const size_t A_OFF    = take((size_t)NK * 128 * 2); // f16 ping / ein (E x 96)
    const size_t B_OFF    = take((size_t)NK * 128 * 2); // f16 pong / eh  (E x 96)
    const size_t C_OFF    = take((size_t)N * 256 * 4);  // fa(NKx64 f16) / aA / g1 / aggr256
    const size_t D_OFF    = take((size_t)N * 256 * 4);  // small bufs / g2 / g3
    const size_t E_OFF    = take((size_t)N * 256 * 4);  // hbuf f32
    const size_t H16_OFF  = take((size_t)N * 256 * 2);  // rotating f16
    const size_t FEAT_OFF = take((size_t)N * 80 * 4);
    const size_t X_OFF    = take((size_t)N * 80 * 4);
    const size_t AG_OFF   = take((size_t)N * 80 * 4);
    const size_t AG2_OFF  = take((size_t)N * 80 * 4);
    const size_t DB_OFF   = take((size_t)NE * 4);
    const size_t PK_OFF   = take((size_t)8 << 20);      // packed-weight arena

    half_t* hA    = (half_t*)(ws + A_OFF);
    half_t* hB    = (half_t*)(ws + B_OFF);
    half_t* h16   = (half_t*)(ws + H16_OFF);
    float*  fC    = (float*)(ws + C_OFF);
    float*  fD    = (float*)(ws + D_OFF);
    float*  fE    = (float*)(ws + E_OFF);
    float*  feat  = (float*)(ws + FEAT_OFF);
    float*  xbuf  = (float*)(ws + X_OFF);
    float*  aggr  = (float*)(ws + AG_OFF);
    float*  aggr2 = (float*)(ws + AG2_OFF);
    float*  dbuf  = (float*)(ws + DB_OFF);

    half_t* fa      = (half_t*)fC;                       // NK x 64
    half_t* aA      = (half_t*)fC;                       // NK x 16 (fa dead)
    half_t* a_nodes = (half_t*)((char*)fD + 0);          // N x 32
    half_t* aB      = (half_t*)((char*)fD + (4 << 20));  // N x 16
    half_t* chem_in = (half_t*)((char*)fD + (8 << 20));  // N x 32
    half_t* c15     = (half_t*)((char*)fD + (12 << 20)); // N x 32
    half_t* ein     = hA;                                // E x 96
    half_t* eh      = hB;                                // E x 96

    // weight packer
    half_t* arena = (half_t*)(ws + PK_OFF);
    size_t pkcur = 0;
    auto pack = [&](int widx, int Kd, int Nout, int Nstore) -> PW {
        int kb = (Kd + 31) / 32, nt = Nstore / 16;
        long total = (long)nt * kb * 512;
        half_t* p = arena + pkcur; pkcur += (size_t)total;
        k_pack_w<<<g1d(total), 256, 0, stream>>>((const float*)d_in[widx], p, Kd, Nout, kb, total);
        return {p, kb};
    };
    auto F = [&](int i) { return (const float*)d_in[i]; };

    // ================= prologue: per-node feature MLPs =================
    k_build_fa<<<g1d(NK * 64), 256, 0, stream>>>(emb, atom_idx, atom_dist, fa, NK * 64);
    gemm(stream, fa, 64,  pack(7, 33, 128, 128),  F(8),  hA, 128, nullptr, (int)NK, 128, 128, true, 0);
    gemm(stream, hA, 128, pack(9, 128, 128, 128), F(10), hB, 128, nullptr, (int)NK, 128, 128, true, 0);
    gemm(stream, hB, 128, pack(11, 128, 128, 128), F(12), hA, 128, nullptr, (int)NK, 128, 128, true, 0);
    gemm(stream, hA, 128, pack(13, 128, 15, 16),  F(14), aA, 16, nullptr, (int)NK, 15, 16, false, 0);
    k_ksum<<<g1d((long)N * 32), 256, 0, stream>>>(aA, a_nodes, (long)N * 32);

    gemm(stream, a_nodes, 32, pack(15, 15, 128, 128), F(16), hA, 128, nullptr, N, 128, 128, true, 0);
    gemm(stream, hA, 128, pack(17, 128, 128, 128), F(18), hB, 128, nullptr, N, 128, 128, true, 0);
    gemm(stream, hB, 128, pack(19, 128, 128, 128), F(20), hA, 128, nullptr, N, 128, 128, true, 0);
    gemm(stream, hA, 128, pack(21, 128, 15, 16),  F(22), aB, 16, nullptr, N, 15, 16, false, 0);

    k_concat16<<<g1d((long)N * 32), 256, 0, stream>>>(aB, node_x, chem_in, (long)N * 32);
    gemm(stream, chem_in, 32, pack(23, 16, 256, 256), F(24), hA, 256, nullptr, N, 256, 256, true, 0);
    gemm(stream, hA, 256, pack(25, 256, 256, 256), F(26), hB, 256, nullptr, N, 256, 256, true, 0);
    gemm(stream, hB, 256, pack(27, 256, 128, 128), F(28), hA, 128, nullptr, N, 128, 128, true, 0);
    gemm(stream, hA, 128, pack(29, 128, 15, 32),  F(30), c15, 32, nullptr, N, 15, 32, false, 0);
    gemm(stream, c15, 32, pack(31, 15, 80, 80),   F(32), feat, 80, nullptr, N, 80, 80, false, 1);

    k_edge_dist<<<g1d(NE), 256, 0, stream>>>(pos, e_src, e_dst, dbuf, NE);

    // ================= 5 edge-MLP + GIN layers =================
    for (int l = 0; l < 5; ++l) {
        const int gb = 61 + 14 * l;

        // ef = MLP([d, edge_attr, feat[dst]]); aef = segment_sum(ef, src) fused in epilogue
        k_build_ein<<<g1d((long)NE * 96), 256, 0, stream>>>(dbuf, edge_attr, feat, e_dst,
                                                            ein, (long)NE * 96);
        gemm(stream, ein, 96, pack(41 + 4*l, 82, 80, 96), F(42 + 4*l),
             eh, 96, nullptr, NE, 80, 96, true, 0);
        k_zero<<<g1d((long)N * 80), 256, 0, stream>>>(aggr, (long)N * 80);
        gemm(stream, eh, 96, pack(43 + 4*l, 80, 80, 80), F(44 + 4*l),
             aggr, 80, e_src, NE, 80, 80, false, 2);

        // x = feat + aef ; hin = x + segment_sum(x[src], dst)
        k_add<<<g1d((long)N * 80), 256, 0, stream>>>(feat, aggr, xbuf, (long)N * 80);
        k_zero<<<g1d((long)N * 80), 256, 0, stream>>>(aggr2, (long)N * 80);
        k_scatter_add<<<g1d((long)NE * 80), 256, 0, stream>>>(xbuf, e_src, e_dst, aggr2,
                                                              80, (long)NE * 80);
        k_add_cvt<<<g1d((long)N * 96), 256, 0, stream>>>(xbuf, aggr2, h16, 80, 96, (long)N * 96);

        // h = relu(ln(hin @ w1a)); h = relu(ln(h @ w1b))
        gemm(stream, h16, 96, pack(gb, 80, 256, 256), F(gb + 1), fC, 256, nullptr, N, 256, 256, false, 1);
        k_ln_relu<<<N, 256, 0, stream>>>(fC, F(gb + 2), F(gb + 3), h16, 256, 256, 1);
        gemm(stream, h16, 256, pack(gb + 4, 256, 256, 256), F(gb + 5), fD, 256, nullptr, N, 256, 256, false, 1);
        k_ln_relu<<<N, 256, 0, stream>>>(fD, F(gb + 6), F(gb + 7), fE, 256, 256, 0);

        // h2 = h + segment_sum(h[src], dst)
        k_zero<<<g1d((long)N * 256), 256, 0, stream>>>(fC, (long)N * 256);
        k_scatter_add<<<g1d((long)NE * 256), 256, 0, stream>>>(fE, e_src, e_dst, fC,
                                                               256, (long)NE * 256);
        k_add_cvt<<<g1d((long)N * 256), 256, 0, stream>>>(fE, fC, h16, 256, 256, (long)N * 256);

        // feat = relu(ln(h2 @ w2a)) @ w2b + b
        gemm(stream, h16, 256, pack(gb + 8, 256, 80, 80), F(gb + 9), fD, 80, nullptr, N, 80, 80, false, 1);
        k_ln_relu<<<N, 256, 0, stream>>>(fD, F(gb + 10), F(gb + 11), h16, 80, 96, 1);
        gemm(stream, h16, 96, pack(gb + 12, 80, 80, 80), F(gb + 13), feat, 80, nullptr, N, 80, 80, false, 1);
    }

    // ================= epilogue: surface mask MLP + outputs =================
    k_cvt_pad<<<g1d((long)N * 96), 256, 0, stream>>>(feat, h16, 80, 96, (long)N * 96);
    gemm(stream, h16, 96, pack(33, 80, 128, 128), F(34), hA, 128, nullptr, N, 128, 128, true, 0);
    gemm(stream, hA, 128, pack(35, 128, 128, 128), F(36), hB, 128, nullptr, N, 128, 128, true, 0);
    gemm(stream, hB, 128, pack(37, 128, 128, 128), F(38), hA, 128, nullptr, N, 128, 128, true, 0);
    gemm(stream, hA, 128, pack(39, 128, 80, 80),  F(40), (float*)d_out, 80, nullptr, N, 80, 80, false, 1);
    k_copy<<<g1d((long)N * 80), 256, 0, stream>>>(feat, (float*)d_out + (size_t)N * 80,
                                                  (long)N * 80);
}
